// FlashAttention2Naive_17617955848637
// MI455X (gfx1250) — matrix operand order repaired and checked by reading, not compiled, not run on hardware
//
#include <hip/hip_runtime.h>
#include <hip/hip_bf16.h>
#include <math.h>

typedef __attribute__((ext_vector_type(16))) _Float16 v16h;
typedef __attribute__((ext_vector_type(8)))  _Float16 v8h;
typedef __attribute__((ext_vector_type(4)))  _Float16 v4h;
typedef __attribute__((ext_vector_type(8)))  float    v8f;

#define SEQ 2048
#define DK  64

__device__ __forceinline__ float rowmax16(float x) {
#pragma unroll
  for (int m = 1; m < 16; m <<= 1) x = fmaxf(x, __shfl_xor(x, m, 32));
  return x;
}
__device__ __forceinline__ float rowsum16(float x) {
#pragma unroll
  for (int m = 1; m < 16; m <<= 1) x += __shfl_xor(x, m, 32);
  return x;
}

// CTA = 256 threads (8 waves) owns a 128-row q tile; all waves share one key
// loop (32 keys/block) so K/V can be staged cooperatively in LDS as f16:
//   Klds[key][d]   -> S  B-operand reads are two contiguous 16B pieces (2x ds_load_b128)
//   Vt[d][key]     -> PV B-operand reads are two contiguous 16B pieces (2x ds_load_b128)
// Each wave computes a 16-row slice: S 16x32 (4 WMMA), O 16x64 (4 WMMA).
__global__ __launch_bounds__(256)
void fa2_fwd_wmma(const float* __restrict__ Q, const float* __restrict__ K,
                  const float* __restrict__ V, float* __restrict__ O,
                  float* __restrict__ L) {
  __shared__ _Float16 Klds[32][64];       // 4 KB: key-major, f16
  __shared__ _Float16 Vt[64][32];         // 4 KB: d-major (transposed), f16
  __shared__ _Float16 p_lds[8][16][32];   // 8 KB: per-wave P re-layout

  const int tid  = threadIdx.x;
  const int lane = tid & 31;
  const int wave = tid >> 5;
  const int lr   = lane & 15;   // column / row-in-half
  const int hl   = lane >> 4;   // 16-lane half

  const int bh     = blockIdx.x >> 4;   // (b*h), 32 total
  const int qtile  = blockIdx.x & 15;   // 16 tiles of 128 rows per (b,h)
  const int q_base = qtile * 128;
  const int q0     = q_base + wave * 16;

  const float scale = 0.125f;  // 1/sqrt(64)

  const float* Qb = Q + (size_t)bh * SEQ * DK;
  const float* Kb = K + (size_t)bh * SEQ * DK;
  const float* Vb = V + (size_t)bh * SEQ * DK;
  float*       Ob = O + (size_t)bh * SEQ * DK;
  float*       Lb = L + (size_t)bh * SEQ;

  // ---- Q tile -> WMMA A-layout (two 16x32 d-chunks), f16 in registers ----
  // A 16x32: lane holds row M=lr; element t -> K = ((t&8)<<1)|(hl<<3)|(t&7)
  v16h qa[2];
  {
    const float* qrow = Qb + (size_t)(q0 + lr) * DK;
#pragma unroll
    for (int c = 0; c < 2; ++c) {
      const int d0 = c * 32 + hl * 8;
      const int d1 = c * 32 + 16 + hl * 8;
#pragma unroll
      for (int t = 0; t < 8; ++t) {
        qa[c][t]     = (_Float16)qrow[d0 + t];
        qa[c][t + 8] = (_Float16)qrow[d1 + t];
      }
    }
  }

  v8f acc[4];
#pragma unroll
  for (int i = 0; i < 4; ++i) acc[i] = v8f{};
  float m_i[8], l_i[8];
#pragma unroll
  for (int v = 0; v < 8; ++v) { m_i[v] = -__builtin_inff(); l_i[v] = 0.0f; }

  const int kend_wave = q0 + 16;                 // this wave's causal bound
  const int nkb       = (q_base + 128) >> 5;     // block-uniform trip count

  for (int kb = 0; kb < nkb; ++kb) {
    const int k0 = kb * 32;

    __syncthreads();  // previous iteration's compute done before overwrite

    // ---- cooperative stage: K block -> Klds (f16), V block -> Vt (f16^T) ----
    // 32x64 f32 = 2048 elems; 256 threads x 2 iters x float4.
#pragma unroll
    for (int r = 0; r < 2; ++r) {
      const int idx = (r * 256 + tid) * 4;   // linear f32 index in block
      const int key = idx >> 6;              // /64
      const int d   = idx & 63;
      const float4 kv = *(const float4*)(Kb + (size_t)(k0 + key) * DK + d);
      v4h kh;
      kh[0] = (_Float16)kv.x; kh[1] = (_Float16)kv.y;
      kh[2] = (_Float16)kv.z; kh[3] = (_Float16)kv.w;
      *(v4h*)&Klds[key][d] = kh;

      const float4 vv = *(const float4*)(Vb + (size_t)(k0 + key) * DK + d);
      Vt[d + 0][key] = (_Float16)vv.x;
      Vt[d + 1][key] = (_Float16)vv.y;
      Vt[d + 2][key] = (_Float16)vv.z;
      Vt[d + 3][key] = (_Float16)vv.w;
    }
    // prefetch next key block into cache hierarchy (global_prefetch_b8)
    if (kb + 1 < nkb && tid < 64) {
      const size_t nxt = (size_t)(k0 + 32) * DK;
      __builtin_prefetch(Kb + nxt + (size_t)tid * 32, 0, 0);
      __builtin_prefetch(Vb + nxt + (size_t)tid * 32, 0, 0);
    }
    __syncthreads();

    if (k0 < kend_wave) {  // wave-uniform causal predicate (EXEC stays full)
      // ---- S(16x32) = Q @ K^T ----
      v8f s[2];
#pragma unroll
      for (int tile = 0; tile < 2; ++tile) {
        v8f c = v8f{};
#pragma unroll
        for (int dc = 0; dc < 2; ++dc) {
          // B 32x16: lane = key column (lr), halfs d = 8*hl+t and 16+8*hl+t
          const v8h blo = *(const v8h*)&Klds[tile * 16 + lr][dc * 32 + hl * 8];
          const v8h bhi = *(const v8h*)&Klds[tile * 16 + lr][dc * 32 + 16 + hl * 8];
          v16h b;
#pragma unroll
          for (int t = 0; t < 8; ++t) { b[t] = blo[t]; b[t + 8] = bhi[t]; }
          c = __builtin_amdgcn_wmma_f32_16x16x32_f16(false, qa[dc], false, b,
                                                     (short)0, c, false, false);
        }
        s[tile] = c;
      }

      // ---- scale + causal mask (diagonal blocks only) ----
      const bool need_mask = (k0 + 31 > q0);
#pragma unroll
      for (int tile = 0; tile < 2; ++tile) {
#pragma unroll
        for (int v = 0; v < 8; ++v) {
          float sv = s[tile][v] * scale;
          if (need_mask) {
            const int key = k0 + tile * 16 + lr;
            const int qr  = q0 + v + hl * 8;
            if (key > qr) sv = -__builtin_inff();
          }
          s[tile][v] = sv;
        }
      }

      // ---- online softmax (row = v + hl*8, across 16 lanes) ----
      float p0[8], p1[8], corr[8];
#pragma unroll
      for (int v = 0; v < 8; ++v) {
        const float mv    = rowmax16(fmaxf(s[0][v], s[1][v]));
        const float m_new = fmaxf(m_i[v], mv);
        const float cr    = __expf(m_i[v] - m_new);
        const float e0    = __expf(s[0][v] - m_new);
        const float e1    = __expf(s[1][v] - m_new);
        const float rs    = rowsum16(e0 + e1);
        l_i[v]  = l_i[v] * cr + rs;
        m_i[v]  = m_new;
        corr[v] = cr;
        p0[v] = e0; p1[v] = e1;
      }
#pragma unroll
      for (int d = 0; d < 4; ++d)
#pragma unroll
        for (int v = 0; v < 8; ++v) acc[d][v] *= corr[v];

      // ---- P: C-layout -> A-layout via per-wave LDS (same-wave DS in order) ----
#pragma unroll
      for (int v = 0; v < 8; ++v) {
        p_lds[wave][v + hl * 8][lr]      = (_Float16)p0[v];
        p_lds[wave][v + hl * 8][16 + lr] = (_Float16)p1[v];
      }
      const v8h plo = *(const v8h*)&p_lds[wave][lr][hl * 8];
      const v8h phi = *(const v8h*)&p_lds[wave][lr][16 + hl * 8];
      v16h pa;
#pragma unroll
      for (int t = 0; t < 8; ++t) { pa[t] = plo[t]; pa[t + 8] = phi[t]; }

      // ---- O(16x64) += P(16x32) @ V(32x64) ----
#pragma unroll
      for (int d = 0; d < 4; ++d) {
        // B 32x16: lane = d column; halfs key = 8*hl+t and 16+8*hl+t (Vt)
        const v8h blo = *(const v8h*)&Vt[d * 16 + lr][hl * 8];
        const v8h bhi = *(const v8h*)&Vt[d * 16 + lr][16 + hl * 8];
        v16h b;
#pragma unroll
        for (int t = 0; t < 8; ++t) { b[t] = blo[t]; b[t + 8] = bhi[t]; }
        acc[d] = __builtin_amdgcn_wmma_f32_16x16x32_f16(false, pa, false, b,
                                                        (short)0, acc[d],
                                                        false, false);
      }
    }
  }

  // ---- epilogue: O = acc / l ; L = m + log(l) ----
#pragma unroll
  for (int v = 0; v < 8; ++v) {
    const float inv_l = 1.0f / l_i[v];
    const int   qr    = q0 + v + hl * 8;
#pragma unroll
    for (int d = 0; d < 4; ++d)
      Ob[(size_t)qr * DK + d * 16 + lr] = acc[d][v] * inv_l;
    if (lr == 0) Lb[qr] = m_i[v] + __logf(l_i[v]);
  }
}

extern "C" void kernel_launch(void* const* d_in, const int* in_sizes, int n_in,
                              void* d_out, int out_size, void* d_ws,
                              size_t ws_size, hipStream_t stream) {
  const float* Q = (const float*)d_in[0];
  const float* K = (const float*)d_in[1];
  const float* V = (const float*)d_in[2];

  const int total = in_sizes[0];          // b*h*seq*dk
  const int BH    = total / (SEQ * DK);   // 32

  float* O = (float*)d_out;                              // O first
  float* L = (float*)d_out + (size_t)BH * SEQ * DK;      // then L

  dim3 grid(BH * (SEQ / 128));            // 8 waves/block, 16 q-rows/wave
  dim3 block(256);
  fa2_fwd_wmma<<<grid, block, 0, stream>>>(Q, K, V, O, L);
}
